// Transformer_88132728914450
// MI455X (gfx1250) — compile-verified, matrix operand order checked by reading, not run on hardware
//
#include <hip/hip_runtime.h>
#include <hip/hip_bf16.h>
#include <math.h>

// ---------------------------------------------------------------------------
// MI455X (gfx1250) transformer forward pass.
// All GEMMs run on v_wmma_f32_16x16x32_bf16 (wave32, 8 waves/block,
// 128x128 block tile, LDS-staged). Attention fused per (batch, head).
// ---------------------------------------------------------------------------

typedef unsigned short u16;
typedef __attribute__((ext_vector_type(16))) __bf16 v16bf;
typedef __attribute__((ext_vector_type(8)))  float  v8f;

constexpr int kD = 1024, kH = 16, kNL = 4, kDFF = 4096, kV = 32000, kB = 16, kS = 128;
constexpr int kM  = kB * kS;   // 2048 tokens
constexpr int kHD = kD / kH;   // 64 head dim

__device__ __forceinline__ u16 f2bf(float f) {
  unsigned u = __float_as_uint(f);
  unsigned r = (u + 0x7FFFu + ((u >> 16) & 1u)) >> 16;  // RNE
  return (u16)r;
}

__device__ __forceinline__ v8f zero8() {
  v8f c;
#pragma unroll
  for (int i = 0; i < 8; ++i) c[i] = 0.0f;
  return c;
}

__device__ __forceinline__ v8f wmma_bf16(v16bf a, v16bf b, v8f c) {
  return __builtin_amdgcn_wmma_f32_16x16x32_bf16(false, a, false, b, (short)0, c,
                                                 false, false);
}

// 16x32 bf16 fragment from a row-major [16, ld] tile at p (element (0,0)).
// ISA A layout: lane L -> row (L&15); lanes 0-15 hold K 0..7 & 16..23,
// lanes 16-31 hold K 8..15 & 24..31. B (from W[N,K] row-major) is symmetric.
__device__ __forceinline__ v16bf load_frag(const u16* __restrict__ p, int ld, int lane) {
  union { v16bf v; uint4 q[2]; } u;
  const u16* r = p + (size_t)(lane & 15) * ld + ((lane >> 4) << 3);
  u.q[0] = *(const uint4*)(r);
  u.q[1] = *(const uint4*)(r + 16);
  return u.v;
}

// ---------------------------------------------------------------------------
// GEMM: out[M,N] = A_bf16[M,K] @ W_bf16[N,K]^T + bias, optional ReLU,
// optional f32 and/or bf16 outputs. Block 256 threads = 8 waves, tile 128x128,
// wave tile 64x32 (4x2 WMMA frags), K step 32, LDS staged.
// ---------------------------------------------------------------------------
template <bool RELU>
__global__ __launch_bounds__(256) void gemm_bf16(
    const u16* __restrict__ A, const u16* __restrict__ W,
    const float* __restrict__ bias, float* __restrict__ outF,
    u16* __restrict__ outB, int M, int N, int K) {
  constexpr int LDP = 40;  // 32 + 8 pad (keeps 16B alignment, spreads banks)
  __shared__ u16 As[128 * LDP];
  __shared__ u16 Bs[128 * LDP];
  const int tid = threadIdx.x, lane = tid & 31, wid = tid >> 5;
  const int bm = blockIdx.y * 128, bn = blockIdx.x * 128;
  const int wm = (wid & 1) * 64, wn = (wid >> 1) * 32;
  const int lr = tid >> 1, lc = (tid & 1) * 16;

  v8f c[4][2];
#pragma unroll
  for (int i = 0; i < 4; ++i)
#pragma unroll
    for (int j = 0; j < 2; ++j) c[i][j] = zero8();

  const u16* Ag = A + (size_t)(bm + lr) * K + lc;
  const u16* Wg = W + (size_t)(bn + lr) * K + lc;

  for (int k0 = 0; k0 < K; k0 += 32) {
    *(uint4*)&As[lr * LDP + lc]     = *(const uint4*)(Ag + k0);
    *(uint4*)&As[lr * LDP + lc + 8] = *(const uint4*)(Ag + k0 + 8);
    *(uint4*)&Bs[lr * LDP + lc]     = *(const uint4*)(Wg + k0);
    *(uint4*)&Bs[lr * LDP + lc + 8] = *(const uint4*)(Wg + k0 + 8);
    if (k0 + 32 < K) {  // global_prefetch_b8 for next K slab
      __builtin_prefetch(Ag + k0 + 32, 0, 1);
      __builtin_prefetch(Wg + k0 + 32, 0, 1);
    }
    __syncthreads();
    v16bf af[4], bfg[2];
#pragma unroll
    for (int i = 0; i < 4; ++i) af[i] = load_frag(&As[(wm + i * 16) * LDP], LDP, lane);
#pragma unroll
    for (int j = 0; j < 2; ++j) bfg[j] = load_frag(&Bs[(wn + j * 16) * LDP], LDP, lane);
#pragma unroll
    for (int i = 0; i < 4; ++i)
#pragma unroll
      for (int j = 0; j < 2; ++j) c[i][j] = wmma_bf16(af[i], bfg[j], c[i][j]);
    __syncthreads();
  }

  // C/D layout: VGPR r -> M = r (lanes 0-15) / 8+r (lanes 16-31); N = lane&15.
  const int hi = (lane >> 4) << 3;
#pragma unroll
  for (int j = 0; j < 2; ++j) {
    const int col = bn + wn + j * 16 + (lane & 15);
    const float bv = bias ? bias[col] : 0.0f;
#pragma unroll
    for (int i = 0; i < 4; ++i) {
      const int rb = bm + wm + i * 16 + hi;
#pragma unroll
      for (int r = 0; r < 8; ++r) {
        float v = c[i][j][r] + bv;
        if (RELU) v = fmaxf(v, 0.0f);
        const size_t off = (size_t)(rb + r) * N + col;
        if (outF) outF[off] = v;
        if (outB) outB[off] = f2bf(v);
      }
    }
  }
}

// ---------------------------------------------------------------------------
// Fused attention for one (batch, head): S=128, head_dim=64.
// scores = (Q K^T)/8 + mask (+causal); softmax; out = P V.
// One block = 8 waves; 64KB LDS score tile; P overwrites scores as bf16.
// ---------------------------------------------------------------------------
__global__ __launch_bounds__(256) void attn_fused(
    const u16* __restrict__ Qb, int q_rs, int q_hs, int q_off,
    const u16* __restrict__ Kb, int k_rs, int k_hs, int k_off,
    const u16* __restrict__ Vt,            // [B*H][64][128] bf16 (V transposed)
    const float* __restrict__ mask,        // [S,S] additive (may be null)
    int causal, u16* __restrict__ Out) {   // [B*S, kD] bf16, merged heads
  __shared__ float Sc[128][128];           // 64 KB
  const int tid = threadIdx.x, lane = tid & 31, wid = tid >> 5;
  const int bh = blockIdx.x, b = bh >> 4, h = bh & 15;
  const u16* Q  = Qb + (size_t)(b * kS) * q_rs + h * q_hs + q_off;
  const u16* Kp = Kb + (size_t)(b * kS) * k_rs + h * k_hs + k_off;
  const int hi = (lane >> 4) << 3;

  {  // ---- scores: wave wid owns query rows [wid*16, wid*16+16), all 128 keys
    v8f c[8];
#pragma unroll
    for (int n = 0; n < 8; ++n) c[n] = zero8();
#pragma unroll
    for (int k0 = 0; k0 < kHD; k0 += 32) {
      v16bf a = load_frag(Q + (size_t)(wid * 16) * q_rs + k0, q_rs, lane);
#pragma unroll
      for (int n = 0; n < 8; ++n) {
        v16bf kb = load_frag(Kp + (size_t)(n * 16) * k_rs + k0, k_rs, lane);
        c[n] = wmma_bf16(a, kb, c[n]);
      }
    }
#pragma unroll
    for (int n = 0; n < 8; ++n)
#pragma unroll
      for (int r = 0; r < 8; ++r)
        Sc[wid * 16 + hi + r][n * 16 + (lane & 15)] = c[n][r] * 0.125f;  // 1/sqrt(64)
  }
  __syncthreads();

  // ---- softmax: one wave per row, lanes stride the 128 columns
  for (int row = wid; row < kS; row += 8) {
    float v[4];
    float mx = -INFINITY;
#pragma unroll
    for (int i = 0; i < 4; ++i) {
      const int j = lane + 32 * i;
      float t = Sc[row][j];
      if (mask) t += mask[row * kS + j];
      if (causal && j > row) t = -INFINITY;
      v[i] = t;
      mx = fmaxf(mx, t);
    }
#pragma unroll
    for (int o = 16; o > 0; o >>= 1) mx = fmaxf(mx, __shfl_xor(mx, o, 32));
    float sum = 0.0f;
#pragma unroll
    for (int i = 0; i < 4; ++i) { v[i] = __expf(v[i] - mx); sum += v[i]; }
#pragma unroll
    for (int o = 16; o > 0; o >>= 1) sum += __shfl_xor(sum, o, 32);
    const float inv = 1.0f / sum;
    u16* Pr = (u16*)&Sc[row][0];  // in-place bf16 P (all reads precede writes)
#pragma unroll
    for (int i = 0; i < 4; ++i) Pr[lane + 32 * i] = f2bf(v[i] * inv);
  }
  __syncthreads();

  // ---- out = P[128,128] @ V[128,64]; B frags from Vt rows (contiguous)
  const u16* P  = (const u16*)&Sc[0][0];       // row stride 256 u16
  const u16* vt = Vt + (size_t)bh * (kHD * kS);
  v8f c2[4];
#pragma unroll
  for (int j = 0; j < 4; ++j) c2[j] = zero8();
#pragma unroll
  for (int k0 = 0; k0 < kS; k0 += 32) {
    v16bf a = load_frag(P + (size_t)(wid * 16) * 256 + k0, 256, lane);
#pragma unroll
    for (int j = 0; j < 4; ++j) {
      v16bf vb = load_frag(vt + (size_t)(j * 16) * kS + k0, kS, lane);
      c2[j] = wmma_bf16(a, vb, c2[j]);
    }
  }
#pragma unroll
  for (int j = 0; j < 4; ++j)
#pragma unroll
    for (int r = 0; r < 8; ++r) {
      const int s = wid * 16 + hi + r;
      const int d = h * kHD + j * 16 + (lane & 15);
      Out[(size_t)(b * kS + s) * kD + d] = f2bf(c2[j][r]);
    }
}

// ---------------------------------------------------------------------------
// Elementwise / helper kernels
// ---------------------------------------------------------------------------
__global__ void cast_bf16(const float* __restrict__ s, u16* __restrict__ d, long long n) {
  long long i = (long long)blockIdx.x * 256 + threadIdx.x;
  if (i < n) d[i] = f2bf(s[i]);
}

// V transpose: vt[(bh*64+d)*128+s] = src[(b*S+s)*rs + h*hs + voff + d]
__global__ void vtrans(const u16* __restrict__ src, int rs, int hs, int voff,
                       u16* __restrict__ vt) {
  int idx = blockIdx.x * 256 + threadIdx.x;
  int s = idx & (kS - 1);
  int t = idx >> 7;
  int d = t & (kHD - 1);
  int bh = t >> 6;
  int b = bh >> 4, h = bh & 15;
  vt[idx] = src[(size_t)(b * kS + s) * rs + h * hs + voff + d];
}

// h0 = (x*emb_w + emb_b)*sqrt(D) + PE; written to encoder AND decoder state.
__global__ void embed_pe(const float* __restrict__ x, const float* __restrict__ ew,
                         const float* __restrict__ ebv, float* __restrict__ ef,
                         u16* __restrict__ ebf, float* __restrict__ yf,
                         u16* __restrict__ ybf) {
  int idx = blockIdx.x * 256 + threadIdx.x;
  int d = idx & (kD - 1);
  int bs = idx >> 10;
  int s = bs & (kS - 1);
  float v = (x[bs] * ew[d] + ebv[d]) * 32.0f;  // sqrt(1024)
  float theta = (float)s * __expf(-((float)(2 * d) / (float)kD) * 9.210340371976184f);
  v += (d & 1) ? __cosf(theta) : __sinf(theta);
  ef[idx] = v; yf[idx] = v;
  u16 bv = f2bf(v);
  ebf[idx] = bv; ybf[idx] = bv;
}

// y = LN(resid + xin)*g + b; one block per token row; writes f32 + bf16.
__global__ __launch_bounds__(256) void ln_residual(
    const float* __restrict__ resid, const float* __restrict__ xin,
    const float* __restrict__ g, const float* __restrict__ bta,
    float* __restrict__ outF, u16* __restrict__ outB) {
  __shared__ float r1[256], r2[256];
  const int row = blockIdx.x, tid = threadIdx.x;
  float vals[4], s = 0.0f, s2 = 0.0f;
#pragma unroll
  for (int i = 0; i < 4; ++i) {
    const int c0 = tid + i * 256;
    float v = resid[(size_t)row * kD + c0] + xin[(size_t)row * kD + c0];
    vals[i] = v; s += v; s2 += v * v;
  }
  r1[tid] = s; r2[tid] = s2;
  __syncthreads();
  for (int o = 128; o > 0; o >>= 1) {
    if (tid < o) { r1[tid] += r1[tid + o]; r2[tid] += r2[tid + o]; }
    __syncthreads();
  }
  const float mean = r1[0] * (1.0f / kD);
  const float var  = r2[0] * (1.0f / kD) - mean * mean;
  const float rstd = rsqrtf(var + 1e-5f);
#pragma unroll
  for (int i = 0; i < 4; ++i) {
    const int c0 = tid + i * 256;
    float y = (vals[i] - mean) * rstd * g[c0] + bta[c0];
    outF[(size_t)row * kD + c0] = y;
    outB[(size_t)row * kD + c0] = f2bf(y);
  }
}

// ---------------------------------------------------------------------------
// Host orchestration
// ---------------------------------------------------------------------------
extern "C" void kernel_launch(void* const* d_in, const int* in_sizes, int n_in,
                              void* d_out, int out_size, void* d_ws, size_t ws_size,
                              hipStream_t stream) {
  (void)in_sizes; (void)n_in; (void)out_size; (void)ws_size;
  const float* x          = (const float*)d_in[0];
  const float* attn_mask  = (const float*)d_in[1];
  const float* emb_w      = (const float*)d_in[2];
  const float* emb_b      = (const float*)d_in[3];
  const float* enc_qkv_w  = (const float*)d_in[4];
  const float* enc_qkv_b  = (const float*)d_in[5];
  const float* enc_out_w  = (const float*)d_in[6];
  const float* enc_out_b  = (const float*)d_in[7];
  const float* enc_ln1_g  = (const float*)d_in[8];
  const float* enc_ln1_b  = (const float*)d_in[9];
  const float* enc_fc1_w  = (const float*)d_in[10];
  const float* enc_fc1_b  = (const float*)d_in[11];
  const float* enc_fc2_w  = (const float*)d_in[12];
  const float* enc_fc2_b  = (const float*)d_in[13];
  const float* enc_ln2_g  = (const float*)d_in[14];
  const float* enc_ln2_b  = (const float*)d_in[15];
  const float* dec_qkv_w  = (const float*)d_in[16];
  const float* dec_qkv_b  = (const float*)d_in[17];
  const float* dec_sa_w   = (const float*)d_in[18];
  const float* dec_sa_b   = (const float*)d_in[19];
  const float* dec_ln1_g  = (const float*)d_in[20];
  const float* dec_ln1_b  = (const float*)d_in[21];
  const float* dec_q_w    = (const float*)d_in[22];
  const float* dec_q_b    = (const float*)d_in[23];
  const float* dec_kv_w   = (const float*)d_in[24];
  const float* dec_kv_b   = (const float*)d_in[25];
  const float* dec_ca_w   = (const float*)d_in[26];
  const float* dec_ca_b   = (const float*)d_in[27];
  const float* dec_ln2_g  = (const float*)d_in[28];
  const float* dec_ln2_b  = (const float*)d_in[29];
  const float* dec_fc1_w  = (const float*)d_in[30];
  const float* dec_fc1_b  = (const float*)d_in[31];
  const float* dec_fc2_w  = (const float*)d_in[32];
  const float* dec_fc2_b  = (const float*)d_in[33];
  const float* dec_ln3_g  = (const float*)d_in[34];
  const float* dec_ln3_b  = (const float*)d_in[35];
  const float* lin_w      = (const float*)d_in[36];
  const float* lin_b      = (const float*)d_in[37];

  // --- bump allocator over d_ws ---
  char* wp = (char*)d_ws;
  auto alloc = [&](size_t bytes) -> void* {
    void* r = (void*)wp;
    wp += (bytes + 255) & ~(size_t)255;
    return r;
  };
  float* ef  = (float*)alloc((size_t)kM * kD * 4);
  float* yf  = (float*)alloc((size_t)kM * kD * 4);
  float* t1f = (float*)alloc((size_t)kM * kD * 4);
  u16* ebf   = (u16*)alloc((size_t)kM * kD * 2);
  u16* ybf   = (u16*)alloc((size_t)kM * kD * 2);
  u16* t0b   = (u16*)alloc((size_t)kM * kDFF * 2);
  u16* attnb = (u16*)alloc((size_t)kM * kD * 2);
  u16* qb    = (u16*)alloc((size_t)kM * kD * 2);
  u16* kvb   = (u16*)alloc((size_t)kM * 2 * kD * 2);
  u16* vtb   = (u16*)alloc((size_t)kM * kD * 2);
  u16* w_enc_qkv = (u16*)alloc((size_t)kNL * 3 * kD * kD * 2);
  u16* w_enc_out = (u16*)alloc((size_t)kNL * kD * kD * 2);
  u16* w_enc_fc1 = (u16*)alloc((size_t)kNL * kDFF * kD * 2);
  u16* w_enc_fc2 = (u16*)alloc((size_t)kNL * kD * kDFF * 2);
  u16* w_dec_qkv = (u16*)alloc((size_t)kNL * 3 * kD * kD * 2);
  u16* w_dec_sa  = (u16*)alloc((size_t)kNL * kD * kD * 2);
  u16* w_dec_q   = (u16*)alloc((size_t)kNL * kD * kD * 2);
  u16* w_dec_kv  = (u16*)alloc((size_t)kNL * 2 * kD * kD * 2);
  u16* w_dec_ca  = (u16*)alloc((size_t)kNL * kD * kD * 2);
  u16* w_dec_fc1 = (u16*)alloc((size_t)kNL * kDFF * kD * 2);
  u16* w_dec_fc2 = (u16*)alloc((size_t)kNL * kD * kDFF * 2);
  u16* w_lin     = (u16*)alloc((size_t)kV * kD * 2);

  auto cast = [&](const float* s, u16* d, long long n) {
    cast_bf16<<<dim3((unsigned)((n + 255) / 256)), dim3(256), 0, stream>>>(s, d, n);
  };
  auto gemm = [&](const u16* A, const u16* W, const float* bias, float* oF, u16* oB,
                  int M, int N, int K, bool relu) {
    dim3 g(N / 128, M / 128);
    if (relu) gemm_bf16<true><<<g, 256, 0, stream>>>(A, W, bias, oF, oB, M, N, K);
    else      gemm_bf16<false><<<g, 256, 0, stream>>>(A, W, bias, oF, oB, M, N, K);
  };
  auto attn = [&](const u16* Qp, int qrs, int qhs, int qoff, const u16* Kp, int krs,
                  int khs, int koff, const u16* Vtp, int causal, u16* Outp) {
    attn_fused<<<dim3(kB * kH), 256, 0, stream>>>(Qp, qrs, qhs, qoff, Kp, krs, khs,
                                                  koff, Vtp, attn_mask, causal, Outp);
  };
  auto vtr = [&](const u16* src, int rs, int hs, int voff) {
    vtrans<<<dim3(kM * kD / 256), 256, 0, stream>>>(src, rs, hs, voff, vtb);
  };

  // --- weight precision conversion (f32 -> bf16) ---
  cast(enc_qkv_w, w_enc_qkv, (long long)kNL * 3 * kD * kD);
  cast(enc_out_w, w_enc_out, (long long)kNL * kD * kD);
  cast(enc_fc1_w, w_enc_fc1, (long long)kNL * kDFF * kD);
  cast(enc_fc2_w, w_enc_fc2, (long long)kNL * kD * kDFF);
  cast(dec_qkv_w, w_dec_qkv, (long long)kNL * 3 * kD * kD);
  cast(dec_sa_w,  w_dec_sa,  (long long)kNL * kD * kD);
  cast(dec_q_w,   w_dec_q,   (long long)kNL * kD * kD);
  cast(dec_kv_w,  w_dec_kv,  (long long)kNL * 2 * kD * kD);
  cast(dec_ca_w,  w_dec_ca,  (long long)kNL * kD * kD);
  cast(dec_fc1_w, w_dec_fc1, (long long)kNL * kDFF * kD);
  cast(dec_fc2_w, w_dec_fc2, (long long)kNL * kD * kDFF);
  cast(lin_w,     w_lin,     (long long)kV * kD);

  // --- embedding + positional encoding (seeds both encoder & decoder state) ---
  embed_pe<<<dim3(kM * kD / 256), 256, 0, stream>>>(x, emb_w, emb_b, ef, ebf, yf, ybf);

  // --- encoder stack ---
  for (int i = 0; i < kNL; ++i) {
    gemm(ebf, w_enc_qkv + (size_t)i * 3 * kD * kD, enc_qkv_b + i * 3 * kD,
         nullptr, t0b, kM, 3 * kD, kD, false);
    // qkv row layout: head h -> q @ h*192, k @ h*192+64, v @ h*192+128
    vtr(t0b, 3 * kD, 3 * kHD, 2 * kHD);
    attn(t0b, 3 * kD, 3 * kHD, 0, t0b, 3 * kD, 3 * kHD, kHD, vtb, 0, attnb);
    gemm(attnb, w_enc_out + (size_t)i * kD * kD, enc_out_b + i * kD,
         t1f, nullptr, kM, kD, kD, false);
    ln_residual<<<dim3(kM), 256, 0, stream>>>(ef, t1f, enc_ln1_g + i * kD,
                                              enc_ln1_b + i * kD, ef, ebf);
    gemm(ebf, w_enc_fc1 + (size_t)i * kDFF * kD, enc_fc1_b + i * kDFF,
         nullptr, t0b, kM, kDFF, kD, true);
    gemm(t0b, w_enc_fc2 + (size_t)i * kD * kDFF, enc_fc2_b + i * kD,
         t1f, nullptr, kM, kD, kDFF, false);
    ln_residual<<<dim3(kM), 256, 0, stream>>>(ef, t1f, enc_ln2_g + i * kD,
                                              enc_ln2_b + i * kD, ef, ebf);
  }

  // --- decoder stack ---
  for (int i = 0; i < kNL; ++i) {
    gemm(ybf, w_dec_qkv + (size_t)i * 3 * kD * kD, dec_qkv_b + i * 3 * kD,
         nullptr, t0b, kM, 3 * kD, kD, false);
    vtr(t0b, 3 * kD, 3 * kHD, 2 * kHD);
    attn(t0b, 3 * kD, 3 * kHD, 0, t0b, 3 * kD, 3 * kHD, kHD, vtb, 1, attnb);  // causal
    gemm(attnb, w_dec_sa + (size_t)i * kD * kD, dec_sa_b + i * kD,
         t1f, nullptr, kM, kD, kD, false);
    ln_residual<<<dim3(kM), 256, 0, stream>>>(yf, t1f, dec_ln1_g + i * kD,
                                              dec_ln1_b + i * kD, yf, ybf);
    // cross attention: Q from decoder, K/V from encoder output
    gemm(ybf, w_dec_q + (size_t)i * kD * kD, dec_q_b + i * kD,
         nullptr, qb, kM, kD, kD, false);
    gemm(ebf, w_dec_kv + (size_t)i * 2 * kD * kD, dec_kv_b + i * 2 * kD,
         nullptr, kvb, kM, 2 * kD, kD, false);
    // kv row layout: head h -> k @ h*128, v @ h*128+64
    vtr(kvb, 2 * kD, 2 * kHD, kHD);
    attn(qb, kD, kHD, 0, kvb, 2 * kD, 2 * kHD, 0, vtb, 0, attnb);
    gemm(attnb, w_dec_ca + (size_t)i * kD * kD, dec_ca_b + i * kD,
         t1f, nullptr, kM, kD, kD, false);
    ln_residual<<<dim3(kM), 256, 0, stream>>>(yf, t1f, dec_ln2_g + i * kD,
                                              dec_ln2_b + i * kD, yf, ybf);
    gemm(ybf, w_dec_fc1 + (size_t)i * kDFF * kD, dec_fc1_b + i * kDFF,
         nullptr, t0b, kM, kDFF, kD, true);
    gemm(t0b, w_dec_fc2 + (size_t)i * kD * kDFF, dec_fc2_b + i * kD,
         t1f, nullptr, kM, kD, kDFF, false);
    ln_residual<<<dim3(kM), 256, 0, stream>>>(yf, t1f, dec_ln3_g + i * kD,
                                              dec_ln3_b + i * kD, yf, ybf);
  }

  // --- vocab projection straight into d_out (f32) ---
  gemm(ybf, w_lin, lin_b, (float*)d_out, nullptr, kM, kV, kD, false);
}
